// MultiHeadAttention_74122545594394
// MI455X (gfx1250) — compile-verified, matrix operand order checked by reading, not run on hardware
//
#include <hip/hip_runtime.h>
#include <hip/hip_bf16.h>

// ---------------------------------------------------------------------------
// MHA forward for MI455X (gfx1250): bf16 WMMA GEMMs + flash attention.
// B=2, T=2048, C=1024, H=16, DK=DV=64.
// ---------------------------------------------------------------------------

constexpr int B_  = 2;
constexpr int T_  = 2048;
constexpr int C_  = 1024;
constexpr int H_  = 16;
constexpr int DK_ = 64;
constexpr int DV_ = 64;

typedef __attribute__((ext_vector_type(16))) __bf16 v16bf;
typedef __attribute__((ext_vector_type(8)))  __bf16 v8bf;
typedef __attribute__((ext_vector_type(4)))  __bf16 v4bf;
typedef __attribute__((ext_vector_type(8)))  float  v8f;
typedef __attribute__((ext_vector_type(4)))  float  v4f;

union BF16Frag { v16bf v; v8bf h[2]; };

__device__ __forceinline__ v8f wmma_bf16(v16bf a, v16bf b, v8f c) {
  // D = A(16x32 bf16) x B(32x16 bf16) + C(16x16 f32)
  return __builtin_amdgcn_wmma_f32_16x16x32_bf16(
      /*neg_a=*/false, a, /*neg_b=*/false, b,
      /*c_mod=*/(short)0, c, /*reuse_a=*/false, /*reuse_b=*/false);
}

// Load one 16(row)x32(K) fragment from row-major bf16 storage with row
// stride `ld` elements. Per ISA 7.12.2: lane L (row = L&15, half = L>>4)
// holds K = 8*half + 0..7 in elems 0..7 and K = 16 + 8*half + 0..7 in 8..15.
__device__ __forceinline__ v16bf load_frag(const __bf16* base, int row, int ld,
                                           int koff, int half) {
  BF16Frag f;
  const __bf16* p = base + row * ld + koff + half * 8;
  f.h[0] = *(const v8bf*)(p);
  f.h[1] = *(const v8bf*)(p + 16);
  return f.v;
}

// ---------------------------------------------------------------------------
// Weight convert+transpose: in f32 [R][Cc] row-major -> out bf16 [Cc][R].
// ---------------------------------------------------------------------------
__global__ __launch_bounds__(256)
void transpose_to_bf16(const float* __restrict__ in, __bf16* __restrict__ out,
                       int R, int Cc) {
  __shared__ float tile[32][33];
  const int tx = threadIdx.x & 31, ty = threadIdx.x >> 5;   // 32 x 8
  const int r0 = blockIdx.y * 32, c0 = blockIdx.x * 32;
#pragma unroll
  for (int j = 0; j < 4; j++)
    tile[ty + j * 8][tx] = in[(size_t)(r0 + ty + j * 8) * Cc + c0 + tx];
  __syncthreads();
#pragma unroll
  for (int j = 0; j < 4; j++)
    out[(size_t)(c0 + ty + j * 8) * R + r0 + tx] = (__bf16)tile[tx][ty + j * 8];
}

// ---------------------------------------------------------------------------
// GEMM: C[M][N] = A[M][K] * B[K][N] + bias[N], Bt pre-transposed bf16 [N][K].
// A f32 (on-the-fly convert) or bf16. 256 threads, 128x128 tile, K-step 32.
// Register double-buffering: next tile's global loads overlap the WMMA block.
// ---------------------------------------------------------------------------
template <bool A_BF16, bool OUT_BF16>
__global__ __launch_bounds__(256)
void gemm_bf16_wmma(const void* __restrict__ Ap, const __bf16* __restrict__ Bt,
                    const float* __restrict__ bias, void* __restrict__ Cp,
                    int M, int N, int K) {
  constexpr int BM = 128, BN = 128, BK = 32, LD = 72;  // 144B rows: 16B aligned
  __shared__ __align__(16) __bf16 As[BM * LD];
  __shared__ __align__(16) __bf16 Bs[BN * LD];
  const int tid  = threadIdx.x;
  const int lane = tid & 31, wid = tid >> 5;
  const int half = lane >> 4, l16 = lane & 15;
  const int wm = wid & 3, wn = wid >> 2;  // 4 waves in M x 2 waves in N
  const int m0 = blockIdx.x * BM, n0 = blockIdx.y * BN;

  v8f zero = {};
  v8f acc[2][4];
#pragma unroll
  for (int i = 0; i < 2; i++)
#pragma unroll
    for (int j = 0; j < 4; j++) acc[i][j] = zero;

  v8bf aRb[2];  v4f aRf[4];  v8bf bR[2];

  auto load_tile = [&](int kk) {
    if constexpr (A_BF16) {
      const __bf16* A = (const __bf16*)Ap;
#pragma unroll
      for (int i = 0; i < 2; i++) {
        const int chunk = tid + i * 256;               // 512 x 16B
        const int r = chunk >> 2, c = (chunk & 3) * 8;
        aRb[i] = *(const v8bf*)(A + (size_t)(m0 + r) * K + kk + c);
      }
    } else {
      const float* A = (const float*)Ap;
#pragma unroll
      for (int i = 0; i < 4; i++) {
        const int chunk = tid + i * 256;               // 1024 x float4
        const int r = chunk >> 3, c = (chunk & 7) * 4;
        aRf[i] = *(const v4f*)(A + (size_t)(m0 + r) * K + kk + c);
      }
    }
#pragma unroll
    for (int i = 0; i < 2; i++) {
      const int chunk = tid + i * 256;
      const int r = chunk >> 2, c = (chunk & 3) * 8;
      bR[i] = *(const v8bf*)(Bt + (size_t)(n0 + r) * K + kk + c);
    }
  };

  auto store_tile = [&]() {
    if constexpr (A_BF16) {
#pragma unroll
      for (int i = 0; i < 2; i++) {
        const int chunk = tid + i * 256;
        const int r = chunk >> 2, c = (chunk & 3) * 8;
        *(v8bf*)(&As[r * LD + c]) = aRb[i];
      }
    } else {
#pragma unroll
      for (int i = 0; i < 4; i++) {
        const int chunk = tid + i * 256;
        const int r = chunk >> 3, c = (chunk & 7) * 4;
        v4bf y;
        y.x = (__bf16)aRf[i].x; y.y = (__bf16)aRf[i].y;
        y.z = (__bf16)aRf[i].z; y.w = (__bf16)aRf[i].w;
        *(v4bf*)(&As[r * LD + c]) = y;
      }
    }
#pragma unroll
    for (int i = 0; i < 2; i++) {
      const int chunk = tid + i * 256;
      const int r = chunk >> 2, c = (chunk & 3) * 8;
      *(v8bf*)(&Bs[r * LD + c]) = bR[i];
    }
  };

  load_tile(0);
  for (int kk = 0; kk < K; kk += BK) {
    __syncthreads();
    store_tile();
    __syncthreads();
    if (kk + BK < K) load_tile(kk + BK);   // overlaps with WMMA block below

    v16bf af[2], bfrag[4];
#pragma unroll
    for (int mi = 0; mi < 2; mi++)
      af[mi] = load_frag(As, wm * 32 + mi * 16 + l16, LD, 0, half);
#pragma unroll
    for (int ni = 0; ni < 4; ni++)
      bfrag[ni] = load_frag(Bs, wn * 64 + ni * 16 + l16, LD, 0, half);
#pragma unroll
    for (int mi = 0; mi < 2; mi++)
#pragma unroll
      for (int ni = 0; ni < 4; ni++)
        acc[mi][ni] = wmma_bf16(af[mi], bfrag[ni], acc[mi][ni]);
  }

  // Epilogue: C/D layout row = v + 8*half, col = l16 (+ tile offsets).
#pragma unroll
  for (int mi = 0; mi < 2; mi++)
#pragma unroll
    for (int ni = 0; ni < 4; ni++) {
      const int col = n0 + wn * 64 + ni * 16 + l16;
      const float bv = bias[col];
#pragma unroll
      for (int v = 0; v < 8; v++) {
        const int row = m0 + wm * 32 + mi * 16 + v + half * 8;
        const float o = acc[mi][ni][v] + bv;
        if constexpr (OUT_BF16)
          ((__bf16*)Cp)[(size_t)row * N + col] = (__bf16)o;
        else
          ((float*)Cp)[(size_t)row * N + col] = o;
      }
    }
}

// ---------------------------------------------------------------------------
// Flash attention, causal. q/k/v bf16 in (B,T,H,D). One block = 64 q-rows of
// one (b,h); 4 waves, each owning 16 q rows. Register-pipelined K/V staging;
// row-max via level-batched bpermute butterfly; row-sum via WMMA x ones.
// Faithful to reference: s = (qk + mask) / sqrt(64), mask added pre-scale.
// ---------------------------------------------------------------------------
__global__ __launch_bounds__(128)
void flash_attn_bf16(const __bf16* __restrict__ qp, const __bf16* __restrict__ kp,
                     const __bf16* __restrict__ vp, __bf16* __restrict__ op) {
  constexpr int LD = 72;
  __shared__ __align__(16) __bf16 Ks[64 * LD];  // [key][d] (B-layout for QK^T)
  __shared__ __align__(16) __bf16 Vs[64 * LD];  // [d][key] (B-layout for P*V)
  __shared__ __align__(16) __bf16 Ps[64 * LD];  // [q][key] (A-layout for P*V)
  const int tid = threadIdx.x, wid = tid >> 5, lane = tid & 31;
  const int half = lane >> 4, l16 = lane & 15;
  const int qt = blockIdx.x, bh = blockIdx.y;
  const int b = bh / H_, h = bh % H_;
  const int qbase = qt * 64;

  // Q fragments: A-layout maps directly onto global (.,H,D) rows.
  const int qrow = qbase + wid * 16 + l16;
  const __bf16* qptr = qp + ((size_t)(b * T_ + qrow) * H_ + h) * DK_;
  BF16Frag aq0, aq1;
  aq0.h[0] = *(const v8bf*)(qptr + half * 8);
  aq0.h[1] = *(const v8bf*)(qptr + half * 8 + 16);
  aq1.h[0] = *(const v8bf*)(qptr + 32 + half * 8);
  aq1.h[1] = *(const v8bf*)(qptr + 32 + half * 8 + 16);

  // Ones B-fragment for row-sum WMMA (all elements 1.0).
  BF16Frag onesf;
#pragma unroll
  for (int i = 0; i < 8; i++) { onesf.h[0][i] = (__bf16)1.0f; onesf.h[1][i] = (__bf16)1.0f; }

  float m_i[8], l_i[8];
  v8f zero = {};
  v8f o[4];
#pragma unroll
  for (int v = 0; v < 8; v++) { m_i[v] = -3.0e38f; l_i[v] = 0.f; }
#pragma unroll
  for (int nd = 0; nd < 4; nd++) o[nd] = zero;

  v8bf kR[4], vR[4];
  auto load_kv = [&](int kbase) {
#pragma unroll
    for (int i = 0; i < 4; i++) {
      const int chunk = tid + i * 128;                 // 512 x 16B
      const int r = chunk >> 3, c = (chunk & 7) * 8;
      kR[i] = *(const v8bf*)(kp + ((size_t)(b * T_ + kbase + r) * H_ + h) * DK_ + c);
      vR[i] = *(const v8bf*)(vp + ((size_t)(b * T_ + kbase + r) * H_ + h) * DV_ + c);
    }
  };
  auto store_kv = [&]() {
#pragma unroll
    for (int i = 0; i < 4; i++) {
      const int chunk = tid + i * 128;
      const int r = chunk >> 3, c = (chunk & 7) * 8;
      *(v8bf*)(&Ks[r * LD + c]) = kR[i];
#pragma unroll
      for (int j = 0; j < 8; j++) Vs[(c + j) * LD + r] = vR[i][j];  // transpose
    }
  };

  load_kv(0);
  for (int kt = 0; kt <= qt; kt++) {
    const int kbase = kt * 64;
    const bool diag = (kt == qt);
    __syncthreads();                 // Ks/Vs consumers of prev iter done
    store_kv();
    __syncthreads();
    if (kt < qt) load_kv(kbase + 64);  // overlaps with compute below

    // S = Q K^T  (16 q-rows x 64 keys per wave; DK=64 => 2 chained WMMAs)
    v8f s[4];
#pragma unroll
    for (int n = 0; n < 4; n++) {
      v16bf bk0 = load_frag(Ks, n * 16 + l16, LD, 0, half);
      v16bf bk1 = load_frag(Ks, n * 16 + l16, LD, 32, half);
      v8f z = zero;
      z = wmma_bf16(aq0.v, bk0, z);
      z = wmma_bf16(aq1.v, bk1, z);
      s[n] = z;
    }
    // Additive causal mask BEFORE scaling (faithful), then 1/sqrt(64).
    if (diag) {
#pragma unroll
      for (int n = 0; n < 4; n++) {
        const int kg = kbase + n * 16 + l16;
#pragma unroll
        for (int v = 0; v < 8; v++) {
          const int qg = qbase + wid * 16 + v + half * 8;
          s[n][v] = (s[n][v] + (kg > qg ? -1e9f : 0.f)) * 0.125f;
        }
      }
    } else {
#pragma unroll
      for (int n = 0; n < 4; n++)
#pragma unroll
        for (int v = 0; v < 8; v++) s[n][v] = s[n][v] * 0.125f;
    }

    // Row max: lane-local over 4 tiles, then level-batched 16-lane butterfly
    // (8 independent bpermutes per level -> one dscnt wait per level).
    float mx[8];
#pragma unroll
    for (int v = 0; v < 8; v++)
      mx[v] = fmaxf(fmaxf(s[0][v], s[1][v]), fmaxf(s[2][v], s[3][v]));
#pragma unroll
    for (int level = 1; level <= 8; level <<= 1) {
      float t[8];
#pragma unroll
      for (int v = 0; v < 8; v++) t[v] = __shfl_xor(mx[v], level);
#pragma unroll
      for (int v = 0; v < 8; v++) mx[v] = fmaxf(mx[v], t[v]);
    }

    float alpha[8];
#pragma unroll
    for (int v = 0; v < 8; v++) {
      const float mnew = fmaxf(m_i[v], mx[v]);
      alpha[v] = __expf(m_i[v] - mnew);
      m_i[v] = mnew;
#pragma unroll
      for (int n = 0; n < 4; n++) s[n][v] = __expf(s[n][v] - mnew);
#pragma unroll
      for (int nd = 0; nd < 4; nd++) o[nd][v] = o[nd][v] * alpha[v];
    }

    // Re-layout P (C-layout regs -> A-layout frags) through LDS.
#pragma unroll
    for (int n = 0; n < 4; n++)
#pragma unroll
      for (int v = 0; v < 8; v++)
        Ps[(wid * 16 + v + half * 8) * LD + n * 16 + l16] = (__bf16)s[n][v];
    __syncthreads();
    v16bf p0 = load_frag(Ps, wid * 16 + l16, LD, 0, half);
    v16bf p1 = load_frag(Ps, wid * 16 + l16, LD, 32, half);

    // Row-sum of P via WMMA against all-ones B: lands replicated in C-layout.
    v8f rs = zero;
    rs = wmma_bf16(p0, onesf.v, rs);
    rs = wmma_bf16(p1, onesf.v, rs);
#pragma unroll
    for (int v = 0; v < 8; v++) l_i[v] = l_i[v] * alpha[v] + rs[v];

#pragma unroll
    for (int nd = 0; nd < 4; nd++) {
      v16bf bv0 = load_frag(Vs, nd * 16 + l16, LD, 0, half);
      v16bf bv1 = load_frag(Vs, nd * 16 + l16, LD, 32, half);
      o[nd] = wmma_bf16(p0, bv0, o[nd]);
      o[nd] = wmma_bf16(p1, bv1, o[nd]);
    }
  }

  // Normalize and store bf16 (B,T,H,DV).
#pragma unroll
  for (int nd = 0; nd < 4; nd++)
#pragma unroll
    for (int v = 0; v < 8; v++) {
      const int qg = qbase + wid * 16 + v + half * 8;
      const int d  = nd * 16 + l16;
      op[((size_t)(b * T_ + qg) * H_ + h) * DV_ + d] = (__bf16)(o[nd][v] / l_i[v]);
    }
}

// ---------------------------------------------------------------------------
// Host launch. Workspace (bf16): 4x 1M weight^T + 4x 4M activations = 40 MB.
// ---------------------------------------------------------------------------
extern "C" void kernel_launch(void* const* d_in, const int* in_sizes, int n_in,
                              void* d_out, int out_size, void* d_ws, size_t ws_size,
                              hipStream_t stream) {
  (void)in_sizes; (void)n_in; (void)out_size; (void)ws_size;
  const float* Q  = (const float*)d_in[0];
  const float* K  = (const float*)d_in[1];
  const float* V  = (const float*)d_in[2];
  const float* Wq = (const float*)d_in[3];
  const float* bq = (const float*)d_in[4];
  const float* Wk = (const float*)d_in[5];
  const float* bk = (const float*)d_in[6];
  const float* Wv = (const float*)d_in[7];
  const float* bv = (const float*)d_in[8];
  const float* Wo = (const float*)d_in[9];
  const float* bo = (const float*)d_in[10];
  // d_in[11] = mask: implemented analytically (causal 0 / -1e9).
  float* out = (float*)d_out;

  __bf16* ws = (__bf16*)d_ws;
  const size_t WSZ = (size_t)1024 * 1024;
  const size_t ASZ = (size_t)(B_ * T_) * 1024;
  __bf16* WtQ = ws;
  __bf16* WtK = WtQ + WSZ;
  __bf16* WtV = WtK + WSZ;
  __bf16* WtO = WtV + WSZ;
  __bf16* qpj = WtO + WSZ;
  __bf16* kpj = qpj + ASZ;
  __bf16* vpj = kpj + ASZ;
  __bf16* apj = vpj + ASZ;

  dim3 tB(256), tG(32, 32);
  transpose_to_bf16<<<tG, tB, 0, stream>>>(Wq, WtQ, 1024, 1024);
  transpose_to_bf16<<<tG, tB, 0, stream>>>(Wk, WtK, 1024, 1024);
  transpose_to_bf16<<<tG, tB, 0, stream>>>(Wv, WtV, 1024, 1024);
  transpose_to_bf16<<<tG, tB, 0, stream>>>(Wo, WtO, 1024, 1024);

  dim3 gB(256), gG((B_ * T_) / 128, 1024 / 128);
  gemm_bf16_wmma<false, true><<<gG, gB, 0, stream>>>(Q, WtQ, bq, qpj, B_ * T_, 1024, 1024);
  gemm_bf16_wmma<false, true><<<gG, gB, 0, stream>>>(K, WtK, bk, kpj, B_ * T_, 1024, 1024);
  gemm_bf16_wmma<false, true><<<gG, gB, 0, stream>>>(V, WtV, bv, vpj, B_ * T_, 1024, 1024);

  flash_attn_bf16<<<dim3(T_ / 64, B_ * H_), dim3(128), 0, stream>>>(qpj, kpj, vpj, apj);

  gemm_bf16_wmma<true, false><<<gG, gB, 0, stream>>>(apj, WtO, bo, out, B_ * T_, 1024, 1024);
}
